// GroupedExperts_8899172237997
// MI455X (gfx1250) — compile-verified, matrix operand order checked by reading, not run on hardware
//
#include <hip/hip_runtime.h>
#include <stdint.h>

// Problem constants (match the reference).
#define T_TOK 8192
#define DDIM  1024
#define HDIM  2816
#define NEXP  8
#define CAPE  1024          // per-expert capacity
#define MT_PER_E (CAPE/128) // 8 M-tiles of 128 rows per expert
#define LDS_STRIDE 48       // shorts per LDS row (32 data + 16 pad, keeps 16B align)

typedef __attribute__((ext_vector_type(16))) __bf16 v16bf;
typedef __attribute__((ext_vector_type(8)))  float  v8f;

__device__ __forceinline__ unsigned short f2bf(float f) {
  unsigned u = __builtin_bit_cast(unsigned, f);
  u += 0x7fffu + ((u >> 16) & 1u);      // round-to-nearest-even
  return (unsigned short)(u >> 16);
}
__device__ __forceinline__ unsigned pack2(float a, float b) {
  return (unsigned)f2bf(a) | ((unsigned)f2bf(b) << 16);
}

union FragU { v16bf v; uint4 q[2]; };

// Build a 16x32 bf16 WMMA operand fragment from LDS.
// lane 0-15: row = base + (lane&15), K = [0,8) U [16,24); lane 16-31: K = [8,16) U [24,32)
// -> matches CDNA5 16-bit A/B VGPR layout.
__device__ __forceinline__ v16bf ldsFrag(const unsigned short* base, int row, int kh) {
  FragU f;
  f.q[0] = *(const uint4*)(base + row * LDS_STRIDE + kh);
  f.q[1] = *(const uint4*)(base + row * LDS_STRIDE + kh + 16);
  return f.v;
}

// Stage a ROWS x 32 tile (corner pointer `src`, row stride `stride` elements)
// into LDS as bf16. float source: convert in-flight. ushort(bf16) source: raw copy.
template <typename T, int ROWS>
__device__ __forceinline__ void stage(const T* src, size_t stride,
                                      unsigned short* lds, int t);

template <int ROWS>
__device__ __forceinline__ void stage_f(const float* src, size_t stride,
                                        unsigned short* lds, int t) {
  #pragma unroll
  for (int i = 0; i < ROWS * 8 / 256; ++i) {
    int c = t + i * 256;
    int r = c >> 3, k4 = (c & 7) * 4;
    const float4 f = *(const float4*)(src + (size_t)r * stride + k4);
    *(unsigned*)(lds + r * LDS_STRIDE + k4)     = pack2(f.x, f.y);
    *(unsigned*)(lds + r * LDS_STRIDE + k4 + 2) = pack2(f.z, f.w);
  }
}
template <int ROWS>
__device__ __forceinline__ void stage_h(const unsigned short* src, size_t stride,
                                        unsigned short* lds, int t) {
  #pragma unroll
  for (int i = 0; i < ROWS * 4 / 256; ++i) {
    int c = t + i * 256;
    int r = c >> 2, k8 = (c & 3) * 8;
    *(uint4*)(lds + r * LDS_STRIDE + k8) =
        *(const uint4*)(src + (size_t)r * stride + k8);
  }
}
template <> __device__ __forceinline__
void stage<float, 128>(const float* s, size_t st, unsigned short* l, int t) { stage_f<128>(s, st, l, t); }
template <> __device__ __forceinline__
void stage<float, 64>(const float* s, size_t st, unsigned short* l, int t) { stage_f<64>(s, st, l, t); }
template <> __device__ __forceinline__
void stage<unsigned short, 128>(const unsigned short* s, size_t st, unsigned short* l, int t) { stage_h<128>(s, st, l, t); }
template <> __device__ __forceinline__
void stage<unsigned short, 64>(const unsigned short* s, size_t st, unsigned short* l, int t) { stage_h<64>(s, st, l, t); }

__device__ __forceinline__ void expert_range(const int* counts, int btile,
                                             int& e, int& row0, int& vrows) {
  e = btile >> 3;
  const int mt = btile & 7;
  int start = 0, cnt = 0;
  #pragma unroll
  for (int i = 0; i < NEXP; ++i) {
    int c = counts[i];
    if (i < e) start += c;
    if (i == e) cnt = c;
  }
  row0  = start + mt * 128;
  vrows = cnt - mt * 128;      // may be <=0 -> tile empty
}

// ---------------- Phase 1: h = silu(x*w1^T) * (x*w3^T)  (bf16 out) -------------
template <typename T>
__global__ __launch_bounds__(256) void moe_h_kernel(
    const T* __restrict__ x, const int* __restrict__ counts,
    const T* __restrict__ w1, const T* __restrict__ w3,
    unsigned short* __restrict__ h) {
  __shared__ unsigned short As[128 * LDS_STRIDE];
  __shared__ unsigned short B1s[64 * LDS_STRIDE];
  __shared__ unsigned short B3s[64 * LDS_STRIDE];

  int e, row0, vrows;
  expert_range(counts, blockIdx.y, e, row0, vrows);
  if (vrows <= 0) return;

  const int ncol0 = blockIdx.x * 64;           // over H
  const int t    = threadIdx.x;
  const int lane = t & 31;
  const int wave = t >> 5;
  const int wm   = wave & 3;                   // 4 waves along M (32 rows each)
  const int wn   = wave >> 2;                  // 2 waves along N (32 cols each)
  const int lm   = lane & 15;
  const int kh   = (lane >> 4) * 8;

  v8f accg[2][2], accu[2][2];
  #pragma unroll
  for (int i = 0; i < 2; ++i)
    #pragma unroll
    for (int j = 0; j < 2; ++j)
      #pragma unroll
      for (int k = 0; k < 8; ++k) { accg[i][j][k] = 0.0f; accu[i][j][k] = 0.0f; }

  const T* xa  = x  + (size_t)row0 * DDIM;
  const T* w1a = w1 + ((size_t)e * HDIM + ncol0) * DDIM;
  const T* w3a = w3 + ((size_t)e * HDIM + ncol0) * DDIM;

  for (int k0 = 0; k0 < DDIM; k0 += 32) {
    __syncthreads();
    stage<T, 128>(xa  + k0, DDIM, As,  t);
    stage<T, 64>( w1a + k0, DDIM, B1s, t);
    stage<T, 64>( w3a + k0, DDIM, B3s, t);
    __syncthreads();

    v16bf a0 = ldsFrag(As,  wm * 32 +  0 + lm, kh);
    v16bf a1 = ldsFrag(As,  wm * 32 + 16 + lm, kh);
    v16bf p0 = ldsFrag(B1s, wn * 32 +  0 + lm, kh);
    v16bf p1 = ldsFrag(B1s, wn * 32 + 16 + lm, kh);
    v16bf q0 = ldsFrag(B3s, wn * 32 +  0 + lm, kh);
    v16bf q1 = ldsFrag(B3s, wn * 32 + 16 + lm, kh);

    accg[0][0] = __builtin_amdgcn_wmma_f32_16x16x32_bf16(false, a0, false, p0, (short)0, accg[0][0], false, false);
    accg[0][1] = __builtin_amdgcn_wmma_f32_16x16x32_bf16(false, a0, false, p1, (short)0, accg[0][1], false, false);
    accg[1][0] = __builtin_amdgcn_wmma_f32_16x16x32_bf16(false, a1, false, p0, (short)0, accg[1][0], false, false);
    accg[1][1] = __builtin_amdgcn_wmma_f32_16x16x32_bf16(false, a1, false, p1, (short)0, accg[1][1], false, false);
    accu[0][0] = __builtin_amdgcn_wmma_f32_16x16x32_bf16(false, a0, false, q0, (short)0, accu[0][0], false, false);
    accu[0][1] = __builtin_amdgcn_wmma_f32_16x16x32_bf16(false, a0, false, q1, (short)0, accu[0][1], false, false);
    accu[1][0] = __builtin_amdgcn_wmma_f32_16x16x32_bf16(false, a1, false, q0, (short)0, accu[1][0], false, false);
    accu[1][1] = __builtin_amdgcn_wmma_f32_16x16x32_bf16(false, a1, false, q1, (short)0, accu[1][1], false, false);
  }

  // Epilogue: h = g * sigmoid(g) * u; v_rcp instead of IEEE divide.
  const int moff = (lane >> 4) * 8;
  #pragma unroll
  for (int mi = 0; mi < 2; ++mi)
    #pragma unroll
    for (int ni = 0; ni < 2; ++ni)
      #pragma unroll
      for (int i = 0; i < 8; ++i) {
        int m = wm * 32 + mi * 16 + moff + i;
        if (m < vrows) {
          float g = accg[mi][ni][i], u = accu[mi][ni][i];
          float s = __builtin_amdgcn_rcpf(1.0f + __expf(-g));
          h[(size_t)(row0 + m) * HDIM + ncol0 + wn * 32 + ni * 16 + lm] =
              f2bf(g * s * u);
        }
      }
}

// ---------------- Phase 2: out = h * w2^T  (fp32 out) --------------------------
template <typename T>
__global__ __launch_bounds__(256) void moe_out_kernel(
    const unsigned short* __restrict__ h, const int* __restrict__ counts,
    const T* __restrict__ w2, float* __restrict__ out) {
  __shared__ unsigned short As[128 * LDS_STRIDE];
  __shared__ unsigned short Bs[64 * LDS_STRIDE];

  int e, row0, vrows;
  expert_range(counts, blockIdx.y, e, row0, vrows);
  if (vrows <= 0) return;

  const int ncol0 = blockIdx.x * 64;           // over D
  const int t    = threadIdx.x;
  const int lane = t & 31;
  const int wave = t >> 5;
  const int wm   = wave & 3;
  const int wn   = wave >> 2;
  const int lm   = lane & 15;
  const int kh   = (lane >> 4) * 8;

  v8f acc[2][2];
  #pragma unroll
  for (int i = 0; i < 2; ++i)
    #pragma unroll
    for (int j = 0; j < 2; ++j)
      #pragma unroll
      for (int k = 0; k < 8; ++k) acc[i][j][k] = 0.0f;

  const unsigned short* ha = h + (size_t)row0 * HDIM;
  const T* w2a = w2 + ((size_t)e * DDIM + ncol0) * HDIM;

  for (int k0 = 0; k0 < HDIM; k0 += 32) {
    __syncthreads();
    stage<unsigned short, 128>(ha + k0, HDIM, As, t);
    stage<T, 64>(w2a + k0, HDIM, Bs, t);
    __syncthreads();

    v16bf a0 = ldsFrag(As, wm * 32 +  0 + lm, kh);
    v16bf a1 = ldsFrag(As, wm * 32 + 16 + lm, kh);
    v16bf b0 = ldsFrag(Bs, wn * 32 +  0 + lm, kh);
    v16bf b1 = ldsFrag(Bs, wn * 32 + 16 + lm, kh);

    acc[0][0] = __builtin_amdgcn_wmma_f32_16x16x32_bf16(false, a0, false, b0, (short)0, acc[0][0], false, false);
    acc[0][1] = __builtin_amdgcn_wmma_f32_16x16x32_bf16(false, a0, false, b1, (short)0, acc[0][1], false, false);
    acc[1][0] = __builtin_amdgcn_wmma_f32_16x16x32_bf16(false, a1, false, b0, (short)0, acc[1][0], false, false);
    acc[1][1] = __builtin_amdgcn_wmma_f32_16x16x32_bf16(false, a1, false, b1, (short)0, acc[1][1], false, false);
  }

  const int moff = (lane >> 4) * 8;
  #pragma unroll
  for (int mi = 0; mi < 2; ++mi)
    #pragma unroll
    for (int ni = 0; ni < 2; ++ni)
      #pragma unroll
      for (int i = 0; i < 8; ++i) {
        int m = wm * 32 + mi * 16 + moff + i;
        if (m < vrows)
          out[(size_t)(row0 + m) * DDIM + ncol0 + wn * 32 + ni * 16 + lm] = acc[mi][ni][i];
      }
}

// ---------------- Helpers: fp32 -> bf16 stream convert, zero-fill --------------
__global__ __launch_bounds__(256) void cvt_bf16_kernel(const float4* __restrict__ src,
                                                       uint2* __restrict__ dst, int n4) {
  int i = blockIdx.x * blockDim.x + threadIdx.x;
  const int stride = gridDim.x * blockDim.x;
  for (; i < n4; i += stride) {
    const float4 f = src[i];
    uint2 o; o.x = pack2(f.x, f.y); o.y = pack2(f.z, f.w);
    dst[i] = o;
  }
}

__global__ __launch_bounds__(256) void zero_out_kernel(float4* __restrict__ p, int n4) {
  int i = blockIdx.x * blockDim.x + threadIdx.x;
  if (i < n4) { float4 z; z.x = z.y = z.z = z.w = 0.0f; p[i] = z; }
}

extern "C" void kernel_launch(void* const* d_in, const int* in_sizes, int n_in,
                              void* d_out, int out_size, void* d_ws, size_t ws_size,
                              hipStream_t stream) {
  const float* x      = (const float*)d_in[0];
  const int*   counts = (const int*)d_in[1];
  const float* w1     = (const float*)d_in[2];
  const float* w2     = (const float*)d_in[3];
  const float* w3     = (const float*)d_in[4];
  float* out = (float*)d_out;

  const size_t hElems = (size_t)T_TOK * HDIM;       // 23.1M
  const size_t xElems = (size_t)T_TOK * DDIM;       // 8.4M
  const size_t wElems = (size_t)NEXP * HDIM * DDIM; // 23.1M per weight
  const size_t need   = (hElems + xElems + 3 * wElems) * sizeof(unsigned short);

  unsigned short* h = (unsigned short*)d_ws;

  const int n4out = T_TOK * DDIM / 4;
  zero_out_kernel<<<(n4out + 255) / 256, 256, 0, stream>>>((float4*)out, n4out);

  const dim3 g1(HDIM / 64, NEXP * MT_PER_E);
  const dim3 g2(DDIM / 64, NEXP * MT_PER_E);

  if (ws_size >= need) {
    // Fast path: pre-convert everything to bf16 once (memory-bound streaming),
    // GEMM staging then does zero-VALU uint4 copies and half the read traffic.
    unsigned short* xb  = h   + hElems;
    unsigned short* w1b = xb  + xElems;
    unsigned short* w3b = w1b + wElems;
    unsigned short* w2b = w3b + wElems;

    cvt_bf16_kernel<<<4096, 256, 0, stream>>>((const float4*)x,  (uint2*)xb,  (int)(xElems / 4));
    cvt_bf16_kernel<<<4096, 256, 0, stream>>>((const float4*)w1, (uint2*)w1b, (int)(wElems / 4));
    cvt_bf16_kernel<<<4096, 256, 0, stream>>>((const float4*)w3, (uint2*)w3b, (int)(wElems / 4));
    cvt_bf16_kernel<<<4096, 256, 0, stream>>>((const float4*)w2, (uint2*)w2b, (int)(wElems / 4));

    moe_h_kernel<unsigned short><<<g1, 256, 0, stream>>>(xb, counts, w1b, w3b, h);
    moe_out_kernel<unsigned short><<<g2, 256, 0, stream>>>(h, counts, w2b, out);
  } else {
    // Fallback: convert fp32 -> bf16 during LDS staging inside the GEMMs.
    moe_h_kernel<float><<<g1, 256, 0, stream>>>(x, counts, w1, w3, h);
    moe_out_kernel<float><<<g2, 256, 0, stream>>>(h, counts, w2, out);
  }
}
